// GraspSampler_2680059593183
// MI455X (gfx1250) — compile-verified, matrix operand order checked by reading, not run on hardware
//
#include <hip/hip_runtime.h>
#include <math.h>

typedef __attribute__((ext_vector_type(16))) _Float16 v16h;
typedef __attribute__((ext_vector_type(8)))  float    v8f;

#define EPSBN 1e-5f

// Folded-BN layer params: y = relu(acc * sc[m] + bi[m])
struct LayerP { const _Float16* w; const float* sc; const float* bi; };

// ---------------------------------------------------------------------------
// Fused GEMM + folded-BN + ReLU over LDS tiles, column-major activations.
// in : LDS f16, column-major [NC cols][CSI halves], channels 0..CINP-1 valid,
//      padding channels CIN..CINP-1 zeroed by the fill loop.
// out: LDS f16, column-major [NC cols][CSO halves]
// W  : global f16 row-major [COUT][CINP]
// CSI/CSO are padded so (stride/2) dwords is odd -> conflict-light LDS.
// Both A and B fragments now load K-contiguous half2 pairs with the same
// index mapping (CDNA5 ISA 7.12.2 16-bit operand layout):
//   half2 index k2 = (q&3) + (q>=4 ? 8 : 0) + hi*4   (K = 2*k2)
// ---------------------------------------------------------------------------
template<int CINP, int COUT, int NC, int CSI, int CSO>
__device__ inline void mlp_layer(const LayerP p, const _Float16* in, _Float16* out)
{
    const int lane = threadIdx.x & 31;
    const int wave = threadIdx.x >> 5;
    const int lm   = lane & 15;
    const int hi   = lane >> 4;
    constexpr int NT  = NC / 16;
    constexpr int WPN = 4 / NT;
    const int nt   = wave % NT;
    const int m0   = wave / NT;
    const int ncol = nt * 16 + lm;
    const unsigned* bcol = (const unsigned*)(in + ncol * CSI);
    unsigned*       ocol = (unsigned*)(out + ncol * CSO);

    for (int mt = m0; mt < COUT / 16; mt += WPN) {
        v8f acc = {};
        const int mrow = mt * 16 + lm;
        const unsigned* arow = (const unsigned*)(p.w + (size_t)mrow * CINP);
        for (int kk = 0; kk < CINP; kk += 32) {
            union { v16h v; unsigned u[8]; } A, Bf;
#pragma unroll
            for (int q = 0; q < 8; ++q) {
                int k2 = (q & 3) + ((q >= 4) ? 8 : 0) + hi * 4;
                A.u[q]  = arow[(kk >> 1) + k2];
                Bf.u[q] = bcol[(kk >> 1) + k2];
            }
            acc = __builtin_amdgcn_wmma_f32_16x16x32_f16(
                false, A.v, false, Bf.v, (short)0, acc, false, false);
        }
        // C/D layout: reg r -> row mt*16 + r + hi*8. Pack row pairs -> b32.
        const int mb = mt * 16 + hi * 8;
#pragma unroll
        for (int i = 0; i < 4; ++i) {
            int m = mb + 2 * i;
            float y0 = fmaxf(acc[2 * i]     * p.sc[m]     + p.bi[m],     0.f);
            float y1 = fmaxf(acc[2 * i + 1] * p.sc[m + 1] + p.bi[m + 1], 0.f);
            union { _Float16 h[2]; unsigned u; } pk;
            pk.h[0] = (_Float16)y0; pk.h[1] = (_Float16)y1;
            ocol[(mb >> 1) + i] = pk.u;
        }
    }
}

// ---------------------------------------------------------------------------
// Weight convert + pad: f32 [cout][cin] -> f16 [cout][cinp], zero padded.
// ---------------------------------------------------------------------------
__global__ void k_cvt_pad(const float* __restrict__ src, _Float16* __restrict__ dst,
                          int cout, int cin, int cinp)
{
    int t = blockIdx.x * 256 + threadIdx.x;
    int total = cout * cinp;
    if (t >= total) return;
    int o = t / cinp, c = t % cinp;
    dst[t] = (c < cin) ? (_Float16)src[o * cin + c] : (_Float16)0.f;
}

// Fold BN (+optional linear bias) into per-channel scale/bias:
//   y = (acc + bias - mean) * g/sqrt(v+eps) + beta  ->  acc*sc + bi
__global__ void k_fold(const float* __restrict__ g, const float* __restrict__ be,
                       const float* __restrict__ mn, const float* __restrict__ vr,
                       const float* __restrict__ bias,
                       float* __restrict__ sc, float* __restrict__ bi, int n)
{
    int t = blockIdx.x * 256 + threadIdx.x;
    if (t >= n) return;
    float s  = g[t] * rsqrtf(vr[t] + EPSBN);
    float b0 = bias ? bias[t] : 0.f;
    sc[t] = s;
    bi[t] = (b0 - mn[t]) * s + be[t];
}

// ---------------------------------------------------------------------------
// Farthest point sampling: one block per batch, shared-mem argmax reduction.
// ---------------------------------------------------------------------------
template<int NSRC, int NP>
__global__ __launch_bounds__(128) void k_fps(const float* __restrict__ pts, int* __restrict__ out)
{
    __shared__ float sd[NSRC];
    __shared__ float rmax[128];
    __shared__ int   rarg[128];
    __shared__ int   slast;
    int b = blockIdx.x, tid = threadIdx.x;
    const float* P = pts + (size_t)b * NSRC * 3;
    for (int j = tid; j < NSRC; j += 128) sd[j] = 1e10f;
    if (tid == 0) { out[(size_t)b * NP] = 0; slast = 0; }
    __syncthreads();
    for (int i = 1; i < NP; ++i) {
        int last = slast;
        float lx = P[last * 3], ly = P[last * 3 + 1], lz = P[last * 3 + 2];
        float best = -1.f; int bj = 0;
        for (int j = tid; j < NSRC; j += 128) {
            float dx = P[j * 3] - lx, dy = P[j * 3 + 1] - ly, dz = P[j * 3 + 2] - lz;
            float d  = dx * dx + dy * dy + dz * dz;
            float nd = fminf(sd[j], d); sd[j] = nd;
            if (nd > best) { best = nd; bj = j; }
        }
        rmax[tid] = best; rarg[tid] = bj;
        __syncthreads();
        for (int off = 64; off; off >>= 1) {
            if (tid < off && rmax[tid + off] > rmax[tid]) {
                rmax[tid] = rmax[tid + off]; rarg[tid] = rarg[tid + off];
            }
            __syncthreads();
        }
        if (tid == 0) { slast = rarg[0]; out[(size_t)b * NP + i] = rarg[0]; }
        __syncthreads();
    }
}

// gather xyz triples by index: dst[t] = src[b][idx[t]]
__global__ void k_gather3(const float* __restrict__ src, const int* __restrict__ idx,
                          float* __restrict__ dst, int nsrc, int S, int total)
{
    int t = blockIdx.x * 256 + threadIdx.x;
    if (t >= total) return;
    int b = t / S;
    int p = idx[t];
    const float* s = src + ((size_t)b * nsrc + p) * 3;
    dst[t * 3 + 0] = s[0]; dst[t * 3 + 1] = s[1]; dst[t * 3 + 2] = s[2];
}

// ---------------------------------------------------------------------------
// Ball query: one wave32 per center; ordered compaction via ballot+popcount.
// ---------------------------------------------------------------------------
template<int NSRC, int NSAMP>
__global__ __launch_bounds__(32) void k_ballq(const float* __restrict__ centers,
                                              const float* __restrict__ pts,
                                              int* __restrict__ out, float r2, int S)
{
    int g = blockIdx.x;
    int b = g / S;
    int lane = threadIdx.x;
    const float* C = centers + (size_t)g * 3;
    float cx = C[0], cy = C[1], cz = C[2];
    const float* P = pts + (size_t)b * NSRC * 3;
    int* O = out + (size_t)g * NSAMP;
    int cnt = 0, firstIdx = -1;
    for (int base = 0; base < NSRC && cnt < NSAMP; base += 32) {
        int j = base + lane;
        bool ok = false;
        if (j < NSRC) {
            float dx = P[j * 3] - cx, dy = P[j * 3 + 1] - cy, dz = P[j * 3 + 2] - cz;
            ok = (dx * dx + dy * dy + dz * dz) <= r2;
        }
        unsigned mm = (unsigned)__ballot(ok);
        if (firstIdx < 0 && mm) firstIdx = base + (__ffs((int)mm) - 1);
        int rank = __popc(mm & ((1u << lane) - 1u));
        if (ok && cnt + rank < NSAMP) O[cnt + rank] = j;
        cnt += __popc(mm);
    }
    if (firstIdx < 0) firstIdx = 0;
    if (cnt > NSAMP) cnt = NSAMP;
    for (int t = cnt + lane; t < NSAMP; t += 32) O[t] = firstIdx;
}

// ---------------------------------------------------------------------------
// SA1 fused: gather(8ch pad->32) -> 32->64->64->128 -> maxpool(64).
// LDS (column-major, halves): sh0 [64][66], sh1 [64][66], sh2 [64][130].
// ---------------------------------------------------------------------------
__global__ __launch_bounds__(128)
void k_sa1(const float* __restrict__ xyz, const float* __restrict__ z,
           const int* __restrict__ fps1, const int* __restrict__ bq1,
           LayerP l0, LayerP l1, LayerP l2, _Float16* __restrict__ pooled1)
{
    extern __shared__ _Float16 smem[];
    _Float16* sh0 = smem;             // 64*66
    _Float16* sh1 = sh0 + 64 * 66;    // 64*66
    _Float16* sh2 = sh1 + 64 * 66;    // 64*130
    int g = blockIdx.x, b = g >> 7, tid = threadIdx.x;
    int ci = fps1[g];
    const float* Cp = xyz + ((size_t)b * 1024 + ci) * 3;
    float cx = Cp[0], cy = Cp[1], cz = Cp[2];
    float z0 = z[b * 2], z1 = z[b * 2 + 1];
    const int* I = bq1 + (size_t)g * 64;
    {   // 2 threads per column, 16 channels each (channels 8..31 zero pad)
        int k = tid >> 1, c0 = (tid & 1) * 16;
        int p = I[k];
        const float* P = xyz + ((size_t)b * 1024 + p) * 3;
        _Float16* col = sh0 + k * 66;
        for (int c = c0; c < c0 + 16; ++c) {
            float v = 0.f;
            if      (c == 0) v = P[0] - cx;
            else if (c == 1) v = P[1] - cy;
            else if (c == 2) v = P[2] - cz;
            else if (c < 6)  v = P[c - 3];
            else if (c < 8)  v = (c == 6) ? z0 : z1;
            col[c] = (_Float16)v;
        }
    }
    __syncthreads();
    mlp_layer<32, 64, 64, 66, 66>(l0, sh0, sh1);   __syncthreads();
    mlp_layer<64, 64, 64, 66, 66>(l1, sh1, sh0);   __syncthreads();
    mlp_layer<64, 128, 64, 66, 130>(l2, sh0, sh2); __syncthreads();
    float m = -3e38f;
    for (int k = 0; k < 64; ++k) m = fmaxf(m, (float)sh2[k * 130 + tid]);
    pooled1[(size_t)g * 128 + tid] = (_Float16)m;
}

// ---------------------------------------------------------------------------
// SA2 fused: gather(131ch pad->160) -> 160->128->128->256 -> maxpool(128).
// 128 cols as 2 chunks of 64. LDS: shIn [64][258] (input uses stride 162),
// shA [64][130], shB [64][130].
// ---------------------------------------------------------------------------
__global__ __launch_bounds__(128)
void k_sa2(const float* __restrict__ nxyz1, const _Float16* __restrict__ pooled1,
           const int* __restrict__ bq2, const float* __restrict__ nxyz2,
           LayerP l0, LayerP l1, LayerP l2, _Float16* __restrict__ pooled2)
{
    extern __shared__ _Float16 smem[];
    _Float16* shIn = smem;               // 64*258
    _Float16* shA  = shIn + 64 * 258;    // 64*130
    _Float16* shB  = shA + 64 * 130;     // 64*130
    int g = blockIdx.x, b = g >> 5, tid = threadIdx.x;
    const float* C = nxyz2 + (size_t)g * 3;
    float cx = C[0], cy = C[1], cz = C[2];
    const int* I = bq2 + (size_t)g * 128;
    float m0 = -3e38f, m1 = -3e38f;
    for (int half = 0; half < 2; ++half) {
        {   // 2 threads per column, 80 channels each (131..159 zero pad)
            int k = tid >> 1, c0 = (tid & 1) * 80;
            int p = I[half * 64 + k];
            const float* P = nxyz1 + ((size_t)b * 128 + p) * 3;
            const _Float16* F = pooled1 + ((size_t)b * 128 + p) * 128;
            _Float16* col = shIn + k * 162;
            for (int c = c0; c < c0 + 80; ++c) {
                _Float16 hv;
                if (c < 3)        hv = (_Float16)(P[c] - (c == 0 ? cx : (c == 1 ? cy : cz)));
                else if (c < 131) hv = F[c - 3];
                else              hv = (_Float16)0.f;
                col[c] = hv;
            }
        }
        __syncthreads();
        mlp_layer<160, 128, 64, 162, 130>(l0, shIn, shA); __syncthreads();
        mlp_layer<128, 128, 64, 130, 130>(l1, shA, shB);  __syncthreads();
        mlp_layer<128, 256, 64, 130, 258>(l2, shB, shIn); __syncthreads();
        for (int k = 0; k < 64; ++k) {
            m0 = fmaxf(m0, (float)shIn[k * 258 + tid]);
            m1 = fmaxf(m1, (float)shIn[k * 258 + tid + 128]);
        }
        __syncthreads();
    }
    pooled2[(size_t)g * 256 + tid]       = (_Float16)m0;
    pooled2[(size_t)g * 256 + tid + 128] = (_Float16)m1;
}

// ---------------------------------------------------------------------------
// SA3 fused (global): 259ch pad->288 -> 288->256->256->512 -> maxpool(32)
// -> xf16 [256 batch][512] (batch-major, FC-ready).
// LDS: shIn [32][514] (input stride 290), shA/shB [32][258].
// ---------------------------------------------------------------------------
__global__ __launch_bounds__(128)
void k_sa3(const float* __restrict__ nxyz2, const _Float16* __restrict__ pooled2,
           LayerP l0, LayerP l1, LayerP l2, _Float16* __restrict__ xf16)
{
    extern __shared__ _Float16 smem[];
    _Float16* shIn = smem;               // 32*514
    _Float16* shA  = shIn + 32 * 514;    // 32*258
    _Float16* shB  = shA + 32 * 258;     // 32*258
    int b = blockIdx.x, tid = threadIdx.x;
    {   // 4 threads per column, 72 channels each (259..287 zero pad)
        int k = tid >> 2, c0 = (tid & 3) * 72;
        const float* P = nxyz2 + ((size_t)b * 32 + k) * 3;
        const _Float16* F = pooled2 + ((size_t)b * 32 + k) * 256;
        _Float16* col = shIn + k * 290;
        for (int c = c0; c < c0 + 72; ++c) {
            _Float16 hv;
            if (c < 3)        hv = (_Float16)P[c];
            else if (c < 259) hv = F[c - 3];
            else              hv = (_Float16)0.f;
            col[c] = hv;
        }
    }
    __syncthreads();
    mlp_layer<288, 256, 32, 290, 258>(l0, shIn, shA); __syncthreads();
    mlp_layer<256, 256, 32, 258, 258>(l1, shA, shB);  __syncthreads();
    mlp_layer<256, 512, 32, 258, 514>(l2, shB, shIn); __syncthreads();
    for (int c = tid; c < 512; c += 128) {
        float m = -3e38f;
        for (int k = 0; k < 32; ++k) m = fmaxf(m, (float)shIn[k * 514 + c]);
        xf16[(size_t)b * 512 + c] = (_Float16)m;
    }
}

// ---------------------------------------------------------------------------
// FC layer (folded BN): out[256][1024] = relu(acc*sc + bi),
// xin batch-major [256][K] so B fragments are 8 b32 global loads (two runs of 4 dwords).
// One block per 16-row m-tile; 4 waves loop the 16 n-tiles (batch columns).
// ---------------------------------------------------------------------------
template<int K>
__global__ __launch_bounds__(128)
void k_fc(LayerP p, const _Float16* __restrict__ xin, _Float16* __restrict__ out)
{
    int mt = blockIdx.x;
    int wave = threadIdx.x >> 5, lane = threadIdx.x & 31;
    int lm = lane & 15, hi = lane >> 4;
    int mrow = mt * 16 + lm;
    const unsigned* arow = (const unsigned*)(p.w + (size_t)mrow * K);
    for (int nt = wave; nt < 16; nt += 4) {
        int ncol = nt * 16 + lm;
        const unsigned* bcol = (const unsigned*)(xin + (size_t)ncol * K);
        v8f acc = {};
        for (int kk = 0; kk < K; kk += 32) {
            union { v16h v; unsigned u[8]; } A, Bf;
#pragma unroll
            for (int q = 0; q < 8; ++q) {
                int k2 = (q & 3) + ((q >= 4) ? 8 : 0) + hi * 4;
                A.u[q]  = arow[(kk >> 1) + k2];
                Bf.u[q] = bcol[(kk >> 1) + k2];
            }
            acc = __builtin_amdgcn_wmma_f32_16x16x32_f16(
                false, A.v, false, Bf.v, (short)0, acc, false, false);
        }
        const int mb = mt * 16 + hi * 8;
        unsigned* ocol = (unsigned*)(out + (size_t)ncol * 1024);
#pragma unroll
        for (int i = 0; i < 4; ++i) {
            int m = mb + 2 * i;
            float y0 = fmaxf(acc[2 * i]     * p.sc[m]     + p.bi[m],     0.f);
            float y1 = fmaxf(acc[2 * i + 1] * p.sc[m + 1] + p.bi[m + 1], 0.f);
            union { _Float16 h[2]; unsigned u; } pk;
            pk.h[0] = (_Float16)y0; pk.h[1] = (_Float16)y1;
            ocol[(mb >> 1) + i] = pk.u;
        }
    }
}

// ---------------------------------------------------------------------------
// Heads: one wave32 per batch row; h2 batch-major [256][1024] -> coalesced.
// ---------------------------------------------------------------------------
__global__ __launch_bounds__(32)
void k_heads(const _Float16* __restrict__ h2,
             const float* __restrict__ qw, const float* __restrict__ qb,
             const float* __restrict__ tw, const float* __restrict__ tb,
             const float* __restrict__ cw, const float* __restrict__ cb,
             float* __restrict__ out)
{
    int b = blockIdx.x, lane = threadIdx.x;
    const _Float16* X = h2 + (size_t)b * 1024;
    float acc[8] = {0, 0, 0, 0, 0, 0, 0, 0};
    for (int j = lane; j < 1024; j += 32) {
        float xv = (float)X[j];
        acc[0] += xv * qw[0 * 1024 + j];
        acc[1] += xv * qw[1 * 1024 + j];
        acc[2] += xv * qw[2 * 1024 + j];
        acc[3] += xv * qw[3 * 1024 + j];
        acc[4] += xv * tw[0 * 1024 + j];
        acc[5] += xv * tw[1 * 1024 + j];
        acc[6] += xv * tw[2 * 1024 + j];
        acc[7] += xv * cw[j];
    }
#pragma unroll
    for (int o = 0; o < 8; ++o)
        for (int off = 16; off; off >>= 1) acc[o] += __shfl_down(acc[o], off, 32);
    if (lane == 0) {
        float q0 = acc[0] + qb[0], q1 = acc[1] + qb[1], q2 = acc[2] + qb[2], q3 = acc[3] + qb[3];
        float nrm = fmaxf(sqrtf(q0 * q0 + q1 * q1 + q2 * q2 + q3 * q3), 1e-12f);
        out[b * 7 + 0] = q0 / nrm; out[b * 7 + 1] = q1 / nrm;
        out[b * 7 + 2] = q2 / nrm; out[b * 7 + 3] = q3 / nrm;
        out[b * 7 + 4] = acc[4] + tb[0];
        out[b * 7 + 5] = acc[5] + tb[1];
        out[b * 7 + 6] = acc[6] + tb[2];
        float cc = acc[7] + cb[0];
        out[256 * 7 + b] = 1.f / (1.f + expf(-cc));
    }
}

// ---------------------------------------------------------------------------
// Host orchestration. Input order = pytree flatten (sorted dict keys):
//  0 xyz, 1 z, 2 c_b, 3 c_w, 4..13 fc{b1,b2,be1,be2,g1,g2,m1,m2,v1,v2},
//  14 fc.w1, 15 fc.w2, 16 q_b, 17 q_w,
//  18..32 sa1 L0..L2 {beta,gamma,mean,var,w}, 33..47 sa2, 48..62 sa3,
//  63 t_b, 64 t_w
// ---------------------------------------------------------------------------
extern "C" void kernel_launch(void* const* d_in, const int* in_sizes, int n_in,
                              void* d_out, int out_size, void* d_ws, size_t ws_size,
                              hipStream_t stream)
{
    (void)in_sizes; (void)n_in; (void)out_size; (void)ws_size;
    const float* xyz = (const float*)d_in[0];
    const float* z   = (const float*)d_in[1];
    auto F = [&](int i) { return (const float*)d_in[i]; };

    char* base = (char*)d_ws;
    size_t ofs = 0;
    auto alloc = [&](size_t bytes) {
        void* p = base + ofs;
        ofs = (ofs + bytes + 255) & ~(size_t)255;
        return p;
    };
    // padded f16 SA weights
    _Float16* wsa[9];
    const int wco[9]  = {64, 64, 128, 128, 128, 256, 256, 256, 512};
    const int wci[9]  = {8, 64, 64, 131, 128, 128, 259, 256, 256};
    const int wcip[9] = {32, 64, 64, 160, 128, 128, 288, 256, 256};
    const int widx[9] = {22, 27, 32, 37, 42, 47, 52, 57, 62};
    for (int i = 0; i < 9; ++i)
        wsa[i] = (_Float16*)alloc((size_t)wco[i] * wcip[i] * 2);
    _Float16* wfc1 = (_Float16*)alloc((size_t)1024 * 512 * 2);
    _Float16* wfc2 = (_Float16*)alloc((size_t)1024 * 1024 * 2);
    // folded BN scale/bias (9 SA layers + 2 FC)
    float* fsc[11]; float* fbi[11];
    const int fn[11] = {64, 64, 128, 128, 128, 256, 256, 256, 512, 1024, 1024};
    for (int i = 0; i < 11; ++i) {
        fsc[i] = (float*)alloc((size_t)fn[i] * 4);
        fbi[i] = (float*)alloc((size_t)fn[i] * 4);
    }
    int*   fps1    = (int*)alloc((size_t)256 * 128 * 4);
    int*   fps2    = (int*)alloc((size_t)256 * 32 * 4);
    float* nxyz1   = (float*)alloc((size_t)256 * 128 * 3 * 4);
    float* nxyz2   = (float*)alloc((size_t)256 * 32 * 3 * 4);
    int*   bq1     = (int*)alloc((size_t)256 * 128 * 64 * 4);
    int*   bq2     = (int*)alloc((size_t)256 * 32 * 128 * 4);
    _Float16* pooled1 = (_Float16*)alloc((size_t)256 * 128 * 128 * 2);
    _Float16* pooled2 = (_Float16*)alloc((size_t)256 * 32 * 256 * 2);
    _Float16* xf16    = (_Float16*)alloc((size_t)256 * 512 * 2);
    _Float16* h1      = (_Float16*)alloc((size_t)256 * 1024 * 2);
    _Float16* h2      = (_Float16*)alloc((size_t)256 * 1024 * 2);

    // ---- weight conversion + BN folding ----
    for (int i = 0; i < 9; ++i) {
        int total = wco[i] * wcip[i];
        k_cvt_pad<<<(total + 255) / 256, 256, 0, stream>>>(F(widx[i]), wsa[i], wco[i], wci[i], wcip[i]);
    }
    k_cvt_pad<<<(1024 * 512 + 255) / 256, 256, 0, stream>>>(F(14), wfc1, 1024, 512, 512);
    k_cvt_pad<<<(1024 * 1024 + 255) / 256, 256, 0, stream>>>(F(15), wfc2, 1024, 1024, 1024);
    // SA layers: params at widx[i]-4 .. widx[i]-1 = beta,gamma,mean,var
    for (int i = 0; i < 9; ++i) {
        int ip = widx[i] - 4;
        k_fold<<<(fn[i] + 255) / 256, 256, 0, stream>>>(
            F(ip + 1), F(ip), F(ip + 2), F(ip + 3), (const float*)nullptr, fsc[i], fbi[i], fn[i]);
    }
    k_fold<<<4, 256, 0, stream>>>(F(8), F(6), F(10), F(12), F(4), fsc[9], fbi[9], 1024);
    k_fold<<<4, 256, 0, stream>>>(F(9), F(7), F(11), F(13), F(5), fsc[10], fbi[10], 1024);

    auto LP = [&](int i) { LayerP p; p.w = wsa[i]; p.sc = fsc[i]; p.bi = fbi[i]; return p; };
    LayerP s1l0 = LP(0), s1l1 = LP(1), s1l2 = LP(2);
    LayerP s2l0 = LP(3), s2l1 = LP(4), s2l2 = LP(5);
    LayerP s3l0 = LP(6), s3l1 = LP(7), s3l2 = LP(8);
    LayerP fc1; fc1.w = wfc1; fc1.sc = fsc[9];  fc1.bi = fbi[9];
    LayerP fc2; fc2.w = wfc2; fc2.sc = fsc[10]; fc2.bi = fbi[10];

    // ---- SA1 ----
    k_fps<1024, 128><<<256, 128, 0, stream>>>(xyz, fps1);
    k_gather3<<<(256 * 128 + 255) / 256, 256, 0, stream>>>(xyz, fps1, nxyz1, 1024, 128, 256 * 128);
    k_ballq<1024, 64><<<256 * 128, 32, 0, stream>>>(nxyz1, xyz, bq1, 0.02f * 0.02f, 128);
    k_sa1<<<256 * 128, 128, (64 * 66 + 64 * 66 + 64 * 130) * 2, stream>>>(
        xyz, z, fps1, bq1, s1l0, s1l1, s1l2, pooled1);

    // ---- SA2 ----
    k_fps<128, 32><<<256, 128, 0, stream>>>(nxyz1, fps2);
    k_gather3<<<(256 * 32 + 255) / 256, 256, 0, stream>>>(nxyz1, fps2, nxyz2, 128, 32, 256 * 32);
    k_ballq<128, 128><<<256 * 32, 32, 0, stream>>>(nxyz2, nxyz1, bq2, 0.04f * 0.04f, 32);
    k_sa2<<<256 * 32, 128, (64 * 258 + 64 * 130 + 64 * 130) * 2, stream>>>(
        nxyz1, pooled1, bq2, nxyz2, s2l0, s2l1, s2l2, pooled2);

    // ---- SA3 ----
    k_sa3<<<256, 128, (32 * 514 + 32 * 258 + 32 * 258) * 2, stream>>>(
        nxyz2, pooled2, s3l0, s3l1, s3l2, xf16);

    // ---- FC + heads ----
    k_fc<512><<<64, 128, 0, stream>>>(fc1, xf16, h1);
    k_fc<1024><<<64, 128, 0, stream>>>(fc2, h1, h2);
    k_heads<<<256, 32, 0, stream>>>(h2, F(17), F(16), F(64), F(63), F(3), F(2),
                                    (float*)d_out);
}
